// DeformableSpatialAttentionLayer_56581899157786
// MI455X (gfx1250) — matrix operand order repaired and checked by reading, not compiled, not run on hardware
//
#include <hip/hip_runtime.h>
#include <math.h>

// ---------------------------------------------------------------------------
// DeformableSpatialAttentionLayer for MI455X (gfx1250, wave32)
//   bs=2, h=w=128 (nq=16384), cin=cout=256, 8 heads x 32 ch, 4 points
// v2: bf16 pre-conversion + TDM (tensor_load_to_lds) double-buffered WMMA GEMM
// Pipeline:
//   P0: convert W_* -> bf16 transposed Wt[n][k]; value/query -> bf16 row-major
//   K1: v      = value @ W_val + b_val          (TDM+WMMA GEMM, head-perm store)
//   K2: offatt = query @ W_off / W_attn + bias  (TDM+WMMA GEMM, 96-wide buffer)
//   K3: softmax + bilinear gather (wave per (q,head)); writes bf16 attnout
//   K4: out    = attnout @ W_out + b_out + query (TDM+WMMA GEMM + residual)
// ---------------------------------------------------------------------------

typedef __bf16 v16bf __attribute__((ext_vector_type(16)));
typedef float  v8f   __attribute__((ext_vector_type(8)));
typedef unsigned int v4u __attribute__((ext_vector_type(4)));
typedef int          v8i __attribute__((ext_vector_type(8)));
typedef int          v4i __attribute__((ext_vector_type(4)));

#define NQ     16384
#define CIN    256
#define HEADS  8
#define NPTS   4
#define DHEAD  32
#define IMGW   128
#define IMGH   128
#define BS     2
#define MTOT   (BS * NQ)

#define TM 128            // block tile M (8 waves x 16 rows)
#define TN 32             // block tile N (two 16-col WMMA tiles per wave)
#define TK 32             // K step (one bf16 WMMA K-depth)
#define LDS_STRIDE 40     // 32 data shorts + 8 pad = 80B (TDM pad: 16dw + 4dw)

// fp32 -> bf16 RNE, scalar
__device__ __forceinline__ unsigned short f2bf(float f) {
  unsigned int u = __float_as_uint(f);
  u += 0x7FFFu + ((u >> 16) & 1u);
  return (unsigned short)(u >> 16);
}
// rounded bits kept in full 32b word (high half = bf16)
__device__ __forceinline__ unsigned int bfr(float f) {
  unsigned int u = __float_as_uint(f);
  return u + 0x7FFFu + ((u >> 16) & 1u);
}
// pack two rounded floats into {bf16(f1), bf16(f0)} via v_perm_b32
__device__ __forceinline__ unsigned int pk2bf(float f0, float f1) {
  return __builtin_amdgcn_perm(bfr(f1), bfr(f0), 0x07060302u);
}

// ---- Tensor Data Mover: 2D bf16 tile (tile_k x rows) global -> LDS --------
// Descriptor bitfields per CDNA5 ISA.
// data_size=1 (2B), pad_enable=1, pad_interval=3 (16 DWORDs), pad_amount=3
// (4 DWORDs) -> LDS row stride = 64B + 16B = 80B = LDS_STRIDE shorts.
__device__ __forceinline__ void tdm_load_tile_bf16(
    unsigned lds_addr, const unsigned short* gptr, int rows, int stride_elems)
{
  unsigned long long ga = (unsigned long long)gptr;
  v4u g0;
  g0[0] = 1u;                                   // count=1, user descriptor
  g0[1] = lds_addr;                             // LDS byte address
  g0[2] = (unsigned)(ga & 0xFFFFFFFFu);         // global_addr[31:0]
  g0[3] = (unsigned)((ga >> 32) & 0x1FFFFFFu)   // global_addr[56:32]
          | (2u << 30);                         // type=2 ("image")
  v8i g1;
  g1[0] = (1 << 16) | (1 << 20) | (3 << 22) | (3 << 25); // dsz=2B, pad 16dw->+4dw
  g1[1] = (TK & 0xFFFF) << 16;                  // tensor_dim0 = TK (lo16)
  g1[2] = (rows & 0xFFFF) << 16;                // dim0 hi=0 | tensor_dim1 = rows
  g1[3] = (TK & 0xFFFF) << 16;                  // dim1 hi=0 | tile_dim0 = TK
  g1[4] = rows & 0xFFFF;                        // tile_dim1 = rows, tile_dim2=0
  g1[5] = stride_elems;                         // tensor_dim0_stride[31:0]
  g1[6] = 0;
  g1[7] = 0;
  v4i g2 = {0, 0, 0, 0};
  v4i g3 = {0, 0, 0, 0};
#if __clang_major__ >= 23
  v8i gz = (v8i)(0);
  __builtin_amdgcn_tensor_load_to_lds(g0, g1, g2, g3, gz, 0);
#else
  __builtin_amdgcn_tensor_load_to_lds(g0, g1, g2, g3, 0);
#endif
}

enum GemmMode { MODE_PLAIN = 0, MODE_VPERM = 1, MODE_RESID = 2 };

// C[m, outcol0+n] = A[m,:] @ Bt[n,:] + bias[n] (+ residual / permuted store)
// A:  M x K bf16 row-major.  Bt: N x K bf16 row-major (pre-transposed W).
template <int MODE>
__global__ __launch_bounds__(256) void gemm_bf16_wmma(
    const unsigned short* __restrict__ A,
    const unsigned short* __restrict__ Bt,
    const float* __restrict__ bias,
    const float* __restrict__ resid,
    float* __restrict__ C,
    int Ksz, int ldc, int outcol0)
{
  __shared__ __align__(16) unsigned short As[2][TM][LDS_STRIDE];
  __shared__ __align__(16) unsigned short Bs[2][TN][LDS_STRIDE];

  const int tid  = threadIdx.x;
  const int wave = tid >> 5;
  const int lane = tid & 31;
  const int m0   = blockIdx.x * TM;
  const int n0b  = blockIdx.y * TN;
  const int nsteps = Ksz / TK;

  // prime stage 0 (TDM issues once per wave; only wave0/lane0 path executes)
  if (tid == 0) {
    tdm_load_tile_bf16((unsigned)(unsigned long long)&As[0][0][0],
                       A + (size_t)m0 * Ksz, TM, Ksz);
    tdm_load_tile_bf16((unsigned)(unsigned long long)&Bs[0][0][0],
                       Bt + (size_t)n0b * Ksz, TN, Ksz);
  }

  v8f acc0 = {};
  v8f acc1 = {};

  for (int s = 0; s < nsteps; ++s) {
    const int buf = s & 1;
    if (tid == 0) {
      if (s + 1 < nsteps) {
        const int k1 = (s + 1) * TK;
        tdm_load_tile_bf16((unsigned)(unsigned long long)&As[buf ^ 1][0][0],
                           A + (size_t)m0 * Ksz + k1, TM, Ksz);
        tdm_load_tile_bf16((unsigned)(unsigned long long)&Bs[buf ^ 1][0][0],
                           Bt + (size_t)n0b * Ksz + k1, TN, Ksz);
        __builtin_amdgcn_s_wait_tensorcnt(2);   // current 2 tiles arrived
      } else {
        __builtin_amdgcn_s_wait_tensorcnt(0);
      }
      asm volatile("" ::: "memory");
    }
    __syncthreads();   // publish TDM-written LDS to all waves

    // ---- WMMA fragments: contiguous k order, same for A and B ----
    // A: lane l -> row (l&15); K chunk 16*(l>>4)..+16 (contiguous)
    // B: lane l -> col (l&15); K chunk 16*(l>>4)..+16 (contiguous)
    const int row  = lane & 15;
    const int aoff = (lane >> 4) * 16;
    const int boff = (lane >> 4) * 16;
    union Frag { v16bf v; float4 f4[2]; } a, b0, b1;
    a.f4[0]  = *(const float4*)&As[buf][wave * 16 + row][aoff];
    a.f4[1]  = *(const float4*)&As[buf][wave * 16 + row][aoff + 8];
    b0.f4[0] = *(const float4*)&Bs[buf][row][boff];
    b0.f4[1] = *(const float4*)&Bs[buf][row][boff + 8];
    b1.f4[0] = *(const float4*)&Bs[buf][16 + row][boff];
    b1.f4[1] = *(const float4*)&Bs[buf][16 + row][boff + 8];

    acc0 = __builtin_amdgcn_wmma_f32_16x16x32_bf16(
        false, a.v, false, b0.v, (short)0, acc0, false, false);
    acc1 = __builtin_amdgcn_wmma_f32_16x16x32_bf16(
        false, a.v, false, b1.v, (short)0, acc1, false, false);

    __syncthreads();   // all waves done with buf before TDM overwrites it
  }

  // ---- epilogue: C/D layout lane l, elem r -> M = r + 8*(l>>4), N = l&15 ----
  const int lhalf = lane >> 4;
  const int ncl   = lane & 15;
#pragma unroll
  for (int r = 0; r < 8; ++r) {
    const int mrow = m0 + wave * 16 + lhalf * 8 + r;
    const int n0w  = n0b + ncl;
    const int n1w  = n0b + 16 + ncl;
    float v0 = acc0[r] + bias[n0w];
    float v1 = acc1[r] + bias[n1w];
    if (MODE == MODE_RESID) {
      v0 += resid[(size_t)mrow * ldc + outcol0 + n0w];
      v1 += resid[(size_t)mrow * ldc + outcol0 + n1w];
    }
    if (MODE == MODE_VPERM) {
      const int b = mrow >> 14;            // / NQ
      const int q = mrow & (NQ - 1);
      C[(((size_t)(b * HEADS + (n0w >> 5))) * NQ + q) * DHEAD + (n0w & 31)] = v0;
      C[(((size_t)(b * HEADS + (n1w >> 5))) * NQ + q) * DHEAD + (n1w & 31)] = v1;
    } else {
      C[(size_t)mrow * ldc + outcol0 + n0w] = v0;
      C[(size_t)mrow * ldc + outcol0 + n1w] = v1;
    }
  }
}

// ---- pre-conversion kernels ------------------------------------------------
// W (K x N fp32) -> Wt (N x K bf16): coalesced reads, tiny matrices
__global__ __launch_bounds__(256) void convert_w_transpose(
    const float* __restrict__ W, unsigned short* __restrict__ Wt, int K, int N)
{
  const int idx = blockIdx.x * 256 + threadIdx.x;
  if (idx >= K * N) return;
  const int k = idx / N, n = idx % N;
  Wt[(size_t)n * K + k] = f2bf(W[idx]);
}

// X (n4 float4) -> Y (n4 uint2 of packed bf16), RNE via v_perm_b32
__global__ __launch_bounds__(256) void convert_act_bf16(
    const float* __restrict__ X, unsigned int* __restrict__ Y, int n4)
{
  const int i = blockIdx.x * 256 + threadIdx.x;
  if (i >= n4) return;
  const float4 f = ((const float4*)X)[i];
  uint2 o;
  o.x = pk2bf(f.x, f.y);
  o.y = pk2bf(f.z, f.w);
  ((uint2*)Y)[i] = o;
}

// ---- sampling: one wave per (b, q, head); lane = channel -------------------
// grid_sample algebra collapses to ix=(q%w)+off_x, iy=(q/w)+off_y.
__global__ __launch_bounds__(256) void deform_sample_kernel(
    const float* __restrict__ v,         // (bs, heads, nq, 32) fp32
    const float* __restrict__ oa,        // (bs*nq, 96): [0..63]=off, [64..95]=attn
    unsigned short* __restrict__ ao)     // (bs*nq, 256) bf16
{
  const int lane = threadIdx.x & 31;
  const int gw   = (blockIdx.x << 3) + (threadIdx.x >> 5);
  const int h    = gw & (HEADS - 1);
  const int mq   = gw >> 3;
  const int q    = mq & (NQ - 1);
  const int b    = mq >> 14;

  const float* po = oa + (size_t)mq * 96;

  float lg[NPTS];
#pragma unroll
  for (int p = 0; p < NPTS; ++p) lg[p] = po[64 + h * NPTS + p];
  float mx = fmaxf(fmaxf(lg[0], lg[1]), fmaxf(lg[2], lg[3]));
  float ex[NPTS];
  float sum = 0.f;
#pragma unroll
  for (int p = 0; p < NPTS; ++p) { ex[p] = __expf(lg[p] - mx); sum += ex[p]; }
  const float inv = 1.0f / sum;

  const float qx = (float)(q & (IMGW - 1));
  const float qy = (float)(q >> 7);
  const float* vb = v + (size_t)(b * HEADS + h) * NQ * DHEAD;

  float acc = 0.f;
#pragma unroll
  for (int p = 0; p < NPTS; ++p) {
    const float ix = qx + po[(h * NPTS + p) * 2 + 0];
    const float iy = qy + po[(h * NPTS + p) * 2 + 1];
    const float x0f = floorf(ix), y0f = floorf(iy);
    const float wx = ix - x0f, wy = iy - y0f;
    const int x0 = (int)x0f, y0 = (int)y0f;
    const float awp = ex[p] * inv;
#pragma unroll
    for (int cyi = 0; cyi < 2; ++cyi) {
      const int yy = y0 + cyi;
      if (yy < 0 || yy >= IMGH) continue;           // wave-uniform branch
      const float wyv = cyi ? wy : (1.f - wy);
#pragma unroll
      for (int cxi = 0; cxi < 2; ++cxi) {
        const int xx = x0 + cxi;
        if (xx < 0 || xx >= IMGW) continue;
        const float wxv = cxi ? wx : (1.f - wx);
        acc += awp * wyv * wxv * vb[((size_t)(yy * IMGW + xx)) * DHEAD + lane];
      }
    }
  }
  ao[(size_t)mq * (HEADS * DHEAD) + h * DHEAD + lane] = f2bf(acc);
}

extern "C" void kernel_launch(void* const* d_in, const int* in_sizes, int n_in,
                              void* d_out, int out_size, void* d_ws, size_t ws_size,
                              hipStream_t stream) {
  const float* query  = (const float*)d_in[0];
  const float* value  = (const float*)d_in[1];
  const float* W_off  = (const float*)d_in[2];
  const float* b_off  = (const float*)d_in[3];
  const float* W_attn = (const float*)d_in[4];
  const float* b_attn = (const float*)d_in[5];
  const float* W_val  = (const float*)d_in[6];
  const float* b_val  = (const float*)d_in[7];
  const float* W_out  = (const float*)d_in[8];
  const float* b_out  = (const float*)d_in[9];
  float* out = (float*)d_out;

  // workspace layout
  float* ws    = (float*)d_ws;
  float* v_ws  = ws;                                        // 8,388,608 f
  float* oa_ws = v_ws + (size_t)BS * HEADS * NQ * DHEAD;    // 3,145,728 f
  unsigned short* us = (unsigned short*)(oa_ws + (size_t)MTOT * 96);
  unsigned short* value_bf = us;                            // MTOT*CIN bf16
  unsigned short* query_bf = value_bf + (size_t)MTOT * CIN;
  unsigned short* ao_bf    = query_bf + (size_t)MTOT * CIN;
  unsigned short* wt_val   = ao_bf + (size_t)MTOT * CIN;    // 256x256
  unsigned short* wt_off   = wt_val + (size_t)CIN * CIN;    // 64x256
  unsigned short* wt_attn  = wt_off + (size_t)64 * CIN;     // 32x256
  unsigned short* wt_out   = wt_attn + (size_t)32 * CIN;    // 256x256

  const dim3 blk(256);
  const int n4act = MTOT * CIN / 4;

  // P0: conversions (weights transposed, activations row-major)
  convert_w_transpose<<<dim3((CIN * CIN + 255) / 256), blk, 0, stream>>>(W_val, wt_val, CIN, CIN);
  convert_w_transpose<<<dim3((CIN * 64 + 255) / 256), blk, 0, stream>>>(W_off, wt_off, CIN, 64);
  convert_w_transpose<<<dim3((CIN * 32 + 255) / 256), blk, 0, stream>>>(W_attn, wt_attn, CIN, 32);
  convert_w_transpose<<<dim3((CIN * CIN + 255) / 256), blk, 0, stream>>>(W_out, wt_out, CIN, CIN);
  convert_act_bf16<<<dim3((n4act + 255) / 256), blk, 0, stream>>>(value, (unsigned int*)value_bf, n4act);
  convert_act_bf16<<<dim3((n4act + 255) / 256), blk, 0, stream>>>(query, (unsigned int*)query_bf, n4act);

  // K1: v projection (head-permuted store)
  gemm_bf16_wmma<MODE_VPERM><<<dim3(MTOT / TM, CIN / TN), blk, 0, stream>>>(
      value_bf, wt_val, b_val, nullptr, v_ws, CIN, 0, 0);
  // K2a: sampling offsets -> oa[:, 0..63]
  gemm_bf16_wmma<MODE_PLAIN><<<dim3(MTOT / TM, 64 / TN), blk, 0, stream>>>(
      query_bf, wt_off, b_off, nullptr, oa_ws, CIN, 96, 0);
  // K2b: attention logits -> oa[:, 64..95]
  gemm_bf16_wmma<MODE_PLAIN><<<dim3(MTOT / TM, 32 / TN), blk, 0, stream>>>(
      query_bf, wt_attn, b_attn, nullptr, oa_ws, CIN, 96, 64);
  // K3: softmax + bilinear gather + weighted sum -> bf16 attnout
  deform_sample_kernel<<<dim3(BS * NQ * HEADS / 8), blk, 0, stream>>>(
      v_ws, oa_ws, ao_bf);
  // K4: output projection + bias + residual(query)
  gemm_bf16_wmma<MODE_RESID><<<dim3(MTOT / TM, CIN / TN), blk, 0, stream>>>(
      ao_bf, wt_out, b_out, query, out, CIN, CIN, 0);
}
